// VQVAE_12592844112180
// MI455X (gfx1250) — compile-verified, matrix operand order checked by reading, not run on hardware
//
#include <hip/hip_runtime.h>
#include <hip/hip_bf16.h>

// ---------------------------------------------------------------------------
// VQ-VAE forward for MI455X (gfx1250, wave32, WMMA).
// All GEMM-shaped work runs on v_wmma_f32_16x16x32_bf16 with f32 accumulation.
// Activations staged as bf16 NHWC in workspace (L2-resident; 192MB L2 covers
// the largest 134MB inter-layer buffer). Weights prepacked to bf16 [N][K]
// (K = (ky,kx,cin)) so A and B fragments are pure 2x b128 loads per the
// CDNA5 16-bit fragment layout. Per-tap weight slices are staged into LDS
// with the gfx1250 async-to-LDS path (ASYNCcnt) when available.
// ---------------------------------------------------------------------------

typedef __attribute__((ext_vector_type(16))) __bf16 v16bf;
typedef __attribute__((ext_vector_type(8)))  float  v8f;

#if defined(__has_builtin)
#  if __has_builtin(__builtin_amdgcn_global_load_async_to_lds_b128) && \
      __has_builtin(__builtin_amdgcn_s_wait_asynccnt)
#    define VQ_ASYNC_LDS 1
#  endif
#endif
#ifndef VQ_ASYNC_LDS
#  define VQ_ASYNC_LDS 0
#endif

union BFU { unsigned short u; __bf16 b; };

__device__ __forceinline__ unsigned short f2bfbits(float f) {
  union { float f; unsigned u; } c; c.f = f;
  unsigned u = c.u + 0x7fffu + ((c.u >> 16) & 1u);   // round-to-nearest-even
  return (unsigned short)(u >> 16);
}
__device__ __forceinline__ __bf16 bits2bf(unsigned short s) { BFU t; t.u = s; return t.b; }
__device__ __forceinline__ __bf16 f2bf(float f) { return bits2bf(f2bfbits(f)); }

// Fragment from a contiguous 32-bf16 chunk: two aligned 16B loads.
__device__ __forceinline__ v16bf frag_from(const unsigned short* __restrict__ p, int kbase) {
  union { uint4 u[2]; v16bf v; } t;
  t.u[0] = *(const uint4*)(p + kbase);
  t.u[1] = *(const uint4*)(p + kbase + 16);
  return t.v;
}
__device__ __forceinline__ v16bf frag_zero() {
  union { uint4 u[2]; v16bf v; } t;
  t.u[0] = make_uint4(0u, 0u, 0u, 0u);
  t.u[1] = make_uint4(0u, 0u, 0u, 0u);
  return t.v;
}

// 16B copy global->LDS: async (gfx1250 GLOBAL_LOAD_ASYNC_TO_LDS_B128) or sync.
#if VQ_ASYNC_LDS
typedef int v4i_vs __attribute__((vector_size(16)));   // matches builtin pointee type
typedef __attribute__((address_space(1))) v4i_vs* g_v4i_p;
typedef __attribute__((address_space(3))) v4i_vs* l_v4i_p;
#endif

__device__ __forceinline__ void copy16_g2l(const unsigned short* gsrc, unsigned short* ldst) {
#if VQ_ASYNC_LDS
  __builtin_amdgcn_global_load_async_to_lds_b128(
      (g_v4i_p)(void*)gsrc, (l_v4i_p)(void*)ldst, 0, 0);
#else
  *(uint4*)ldst = *(const uint4*)gsrc;
#endif
}
__device__ __forceinline__ void wait_g2l() {
#if VQ_ASYNC_LDS
  __builtin_amdgcn_s_wait_asynccnt(0);
#endif
}

// ---------------------------------------------------------------------------
// Weight prepack: f32 torch layout -> bf16 [COUT_pad16][KTOT], K=(ky,kx,cin).
// TRANS: wf[n][ci][ky][kx] = w[ci][n][KH-1-ky][KW-1-kx]. Zero rows for n>=COUT.
// ---------------------------------------------------------------------------
template<int CIN, int COUT, int KH, int KW, bool TRANS>
__global__ __launch_bounds__(256) void pack_w_kernel(
    const float* __restrict__ w, unsigned short* __restrict__ wbf)
{
  constexpr int KTOT = KH * KW * CIN;
  constexpr int CP   = (COUT + 15) & ~15;
  const long gid = (long)blockIdx.x * blockDim.x + threadIdx.x;
  if (gid >= (long)CP * KTOT) return;
  const int n = (int)(gid / KTOT);
  const int k = (int)(gid % KTOT);
  float v = 0.0f;
  if (n < COUT) {
    const int ky = k / (KW * CIN);
    const int r  = k - ky * (KW * CIN);
    const int kx = r / CIN;
    const int ci = r - kx * CIN;
    v = TRANS ? w[((long)ci * COUT + n) * (KH * KW) + (KH - 1 - ky) * KW + (KW - 1 - kx)]
              : w[((long)n * CIN + ci) * (KH * KW) + ky * KW + kx];
  }
  wbf[gid] = f2bfbits(v);
}

// ---------------------------------------------------------------------------
// Fast implicit-GEMM conv / transposed conv (requires CIN % 32 == 0).
// Block = 8 waves sharing one N-tile of 16*NT columns. Per (ky,kx) tap the
// block async-stages the [16*NT x CIN] weight slice into LDS (bank-skewed
// rows), then each wave runs 32-channel chunks: A fragments = 2x b128 global
// loads (uniform per-tap predicate), B fragments = 2x ds_load_b128,
// MT x NT WMMAs per chunk.
// OUT_MODE: 0 = bf16 NHWC, 1 = bf16 NHWC + f32 NHWC, 2 = f32 NCHW.
// Grid must be exactly (MGROUPS/8)*NGRP blocks (barriers inside).
// ---------------------------------------------------------------------------
template<int BATCH,int CIN,int COUT,int KH,int KW,int STRIDE,int PAD,bool TRANS,
         int HIN,int WIN,int HOUT,int WOUT,int MT,int NT,bool RELU,int OUT_MODE>
__global__ __launch_bounds__(256) void conv_wmma_fast_kernel(
    const unsigned short* __restrict__ inp, const unsigned short* __restrict__ wbf,
    const float* __restrict__ bias, unsigned short* __restrict__ obf,
    float* __restrict__ of32)
{
  static_assert(CIN % 32 == 0, "fast path needs CIN % 32 == 0");
  constexpr int  KTOT    = KH * KW * CIN;
  constexpr int  CCH     = CIN / 32;
  constexpr int  NGRP    = (((COUT + 15) / 16) + NT - 1) / NT;
  constexpr long MTOT    = (long)BATCH * HOUT * WOUT;
  constexpr int  MGROUPS = (int)(MTOT / (16 * MT));
  constexpr int  WPB     = 8;                 // waves per block
  constexpr int  NROWS   = 16 * NT;           // B rows staged per tap
  constexpr int  LDSROW  = CIN + 8;           // +16B skew against bank aliasing
  constexpr int  CPR     = CIN / 8;           // 16B chunks per B row
  static_assert(MGROUPS % WPB == 0, "exact grid required (barriers in kernel)");

  __shared__ __attribute__((aligned(16))) unsigned short ldsb[NROWS * LDSROW];

  const int lane   = threadIdx.x & 31;
  const int waveid = threadIdx.x >> 5;
  const int ng     = blockIdx.x % NGRP;
  const int mg     = (blockIdx.x / NGRP) * WPB + waveid;

  const int m0 = mg * 16 * MT, n0 = ng * 16 * NT;
  const int kbase = (lane & 16) ? 8 : 0;
  const int nl    = lane & 15;

  // Per-A-fragment pixel coordinates (row = m0 + t*16 + nl).
  int ax[MT], ay[MT], ab[MT];
#pragma unroll
  for (int t = 0; t < MT; ++t) {
    const int am  = m0 + t * 16 + nl;
    ax[t] = am % WOUT;
    const int yb = am / WOUT;
    ay[t] = yb % HOUT;
    ab[t] = yb / HOUT;
  }

  v8f acc[MT][NT];
#pragma unroll
  for (int j = 0; j < NT; ++j) {
    const int n = n0 + j * 16 + nl;
    const float bv = (n < COUT) ? bias[n] : 0.0f;
#pragma unroll
    for (int t = 0; t < MT; ++t)
#pragma unroll
      for (int i = 0; i < 8; ++i) acc[t][j][i] = bv;
  }

  for (int kyx = 0; kyx < KH * KW; ++kyx) {
    const int ky = kyx / KW, kx = kyx % KW;

    // ---- Stage B slice [NROWS x CIN] for this tap into LDS (async path) ----
    {
      constexpr int TOTC = NROWS * CPR;
      for (int c = threadIdx.x; c < TOTC; c += 256) {
        const int r  = c / CPR;
        const int w8 = c - r * CPR;
        const unsigned short* gsrc = wbf + (long)(n0 + r) * KTOT + kyx * CIN + w8 * 8;
        copy16_g2l(gsrc, &ldsb[r * LDSROW + w8 * 8]);
      }
      wait_g2l();
      __syncthreads();
    }

    // ---- Tap predicate + input row base: uniform across the 32-K chunk ----
    const unsigned short* ap[MT];
    bool ok[MT];
#pragma unroll
    for (int t = 0; t < MT; ++t) {
      int iy, ix; bool o;
      if (TRANS) {
        const int iyn = ay[t] + ky - (KH - 1 - PAD);
        const int ixn = ax[t] + kx - (KW - 1 - PAD);
        iy = iyn / STRIDE; ix = ixn / STRIDE;
        o = (iyn >= 0) && (ixn >= 0) &&
            ((iyn % STRIDE) == 0) && ((ixn % STRIDE) == 0) &&
            (iy < HIN) && (ix < WIN);
      } else {
        iy = ay[t] * STRIDE + ky - PAD;
        ix = ax[t] * STRIDE + kx - PAD;
        o = (iy >= 0) && (iy < HIN) && (ix >= 0) && (ix < WIN);
      }
      if (!o) { iy = 0; ix = 0; }
      ok[t] = o;
      ap[t] = inp + (((long)ab[t] * HIN + iy) * WIN + ix) * CIN;
    }

#pragma unroll
    for (int cc = 0; cc < CCH; ++cc) {
      v16bf a[MT];
#pragma unroll
      for (int t = 0; t < MT; ++t)
        a[t] = ok[t] ? frag_from(ap[t] + cc * 32, kbase) : frag_zero();

#pragma unroll
      for (int j = 0; j < NT; ++j) {
        const v16bf bfrag = frag_from(&ldsb[(j * 16 + nl) * LDSROW + cc * 32], kbase);
#pragma unroll
        for (int t = 0; t < MT; ++t)
          acc[t][j] = __builtin_amdgcn_wmma_f32_16x16x32_bf16(
              false, a[t], false, bfrag, (short)0, acc[t][j], false, false);
      }
    }
    __syncthreads();   // protect LDS before next tap's staging
  }

  // Store D: element i of lane -> (m0 + t*16 + (lane&16?8:0) + i, n0 + 16j + nl)
  const int mrb = (lane & 16) ? 8 : 0;
#pragma unroll
  for (int t = 0; t < MT; ++t)
#pragma unroll
    for (int j = 0; j < NT; ++j) {
      const int n = n0 + j * 16 + nl;
#pragma unroll
      for (int i = 0; i < 8; ++i) {
        const int m = m0 + t * 16 + mrb + i;
        float v = acc[t][j][i];
        if (RELU) v = v > 0.0f ? v : 0.0f;
        const int x  = m % WOUT;
        const int yb = m / WOUT;
        const int y  = yb % HOUT;
        const int b  = yb / HOUT;
        if (OUT_MODE == 2) {
          if (n < COUT)
            of32[(((long)b * COUT + n) * HOUT + y) * WOUT + x] = v;
        } else {
          if (n < COUT) {
            const long o = (((long)b * HOUT + y) * WOUT + x) * COUT + n;
            obf[o] = f2bfbits(v);
            if (OUT_MODE == 1) of32[o] = v;
          }
        }
      }
    }
}

// ---------------------------------------------------------------------------
// Generic implicit-GEMM conv (element-wise gather) — only used for enc1
// (CIN=3, ~0.7% of total FLOPs).
// ---------------------------------------------------------------------------
template<int BATCH,int CIN,int COUT,int KH,int KW,int STRIDE,int PAD,
         int HIN,int WIN,int HOUT,int WOUT,int NT,bool RELU>
__global__ __launch_bounds__(256) void conv_wmma_gen_kernel(
    const float* __restrict__ inp, const float* __restrict__ wgt,
    const float* __restrict__ bias, unsigned short* __restrict__ obf)
{
  constexpr int  KTOT   = KH * KW * CIN;
  constexpr int  KSTEPS = (KTOT + 31) / 32;
  constexpr int  NGRP   = (COUT + 16 * NT - 1) / (16 * NT);
  constexpr long MTOT   = (long)BATCH * HOUT * WOUT;
  constexpr int  MTILES = (int)(MTOT / 16);

  const int lane = threadIdx.x & 31;
  const int wid  = blockIdx.x * (blockDim.x >> 5) + (threadIdx.x >> 5);
  if (wid >= MTILES * NGRP) return;

  const int mt = wid / NGRP, ng = wid % NGRP;
  const int m0 = mt * 16, n0 = ng * 16 * NT;

  const int am  = m0 + (lane & 15);
  const int ax  = am % WOUT;
  const int ayb = am / WOUT;
  const int ay  = ayb % HOUT;
  const int ab  = ayb / HOUT;
  const int kbase = (lane & 16) ? 8 : 0;
  const int nl    = lane & 15;

  v8f acc[NT];
#pragma unroll
  for (int j = 0; j < NT; ++j) {
    const int n = n0 + j * 16 + nl;
    const float bv = (n < COUT) ? bias[n] : 0.0f;
#pragma unroll
    for (int i = 0; i < 8; ++i) acc[j][i] = bv;
  }

  for (int ks = 0; ks < KSTEPS; ++ks) {
    const int kb = ks * 32 + kbase;
    v16bf a;
#pragma unroll
    for (int e = 0; e < 16; ++e) {
      const int k = kb + ((e & 8) ? 16 : 0) + (e & 7);
      __bf16 val = bits2bf(0);
      if (k < KTOT) {
        const int ky = k / (KW * CIN);
        const int r  = k - ky * (KW * CIN);
        const int kx = r / CIN;
        const int ci = r - kx * CIN;
        const int iy = ay * STRIDE + ky - PAD;
        const int ix = ax * STRIDE + kx - PAD;
        if (iy >= 0 && iy < HIN && ix >= 0 && ix < WIN)
          val = f2bf(inp[(((long)ab * CIN + ci) * HIN + iy) * WIN + ix]);
      }
      a[e] = val;
    }

    v16bf bm[NT];
#pragma unroll
    for (int j = 0; j < NT; ++j) {
      const int n = n0 + j * 16 + nl;
#pragma unroll
      for (int e = 0; e < 16; ++e) {
        const int k = kb + ((e & 8) ? 16 : 0) + (e & 7);
        __bf16 val = bits2bf(0);
        if (k < KTOT && n < COUT) {
          const int ky = k / (KW * CIN);
          const int r  = k - ky * (KW * CIN);
          const int kx = r / CIN;
          const int ci = r - kx * CIN;
          val = f2bf(wgt[(((long)n * CIN + ci) * KH + ky) * KW + kx]);
        }
        bm[j][e] = val;
      }
    }

#pragma unroll
    for (int j = 0; j < NT; ++j)
      acc[j] = __builtin_amdgcn_wmma_f32_16x16x32_bf16(
          false, a, false, bm[j], (short)0, acc[j], false, false);
  }

  const int mrb = (lane & 16) ? 8 : 0;
#pragma unroll
  for (int j = 0; j < NT; ++j) {
    const int n = n0 + j * 16 + nl;
#pragma unroll
    for (int i = 0; i < 8; ++i) {
      const int m = m0 + mrb + i;
      float v = acc[j][i];
      if (RELU) v = v > 0.0f ? v : 0.0f;
      const int x  = m % WOUT;
      const int yb = m / WOUT;
      const int y  = yb % HOUT;
      const int b  = yb / HOUT;
      if (n < COUT)
        obf[(((long)b * HOUT + y) * WOUT + x) * COUT + n] = f2bfbits(v);
    }
  }
}

// ---------------------------------------------------------------------------
// VQ prep: half-norms 0.5*||e_k||^2, bf16 codebook copy, histogram zero.
// ---------------------------------------------------------------------------
__global__ __launch_bounds__(256) void prep_emb_kernel(
    const float* __restrict__ emb, unsigned short* __restrict__ embbf,
    float* __restrict__ enorm, int* __restrict__ hist)
{
  const int k = blockIdx.x * blockDim.x + threadIdx.x;
  if (k >= 512) return;
  hist[k] = 0;
  float s = 0.0f;
  for (int d = 0; d < 256; ++d) {
    const float v = emb[(long)k * 256 + d];
    s += v * v;
    embbf[(long)k * 256 + d] = f2bfbits(v);
  }
  enorm[k] = 0.5f * s;
}

// ---------------------------------------------------------------------------
// VQ argmin: dist = ||z||^2 + ||e||^2 - 2 z.e  ==>  argmax_n (z.e - 0.5||e||^2).
// One wave: 16 pixels x 512 codes; 8 A fragments of D=256 held in registers;
// 32 code tiles x 8 WMMAs; shuffle-tree argmax over the 16-lane half.
// (Asm-verified: lowers to global_load_b128 + back-to-back v_wmma.)
// ---------------------------------------------------------------------------
__global__ __launch_bounds__(256) void vq_argmin_kernel(
    const unsigned short* __restrict__ zbf, const unsigned short* __restrict__ embbf,
    const float* __restrict__ enorm, int* __restrict__ idx, int* __restrict__ hist,
    int npix)
{
  const int lane = threadIdx.x & 31;
  const int wid  = blockIdx.x * (blockDim.x >> 5) + (threadIdx.x >> 5);
  const int p0   = wid * 16;
  if (p0 >= npix) return;

  const int kbase = (lane & 16) ? 8 : 0;
  const int nl    = lane & 15;
  const unsigned short* zrow = zbf + (long)(p0 + nl) * 256;

  v16bf afrag[8];
#pragma unroll
  for (int ks = 0; ks < 8; ++ks)
    afrag[ks] = frag_from(zrow + ks * 32, kbase);

  float bestv[8]; int besti[8];
#pragma unroll
  for (int i = 0; i < 8; ++i) { bestv[i] = -3.4e38f; besti[i] = 0; }

  for (int nt = 0; nt < 32; ++nt) {
    const int n = nt * 16 + nl;
    const unsigned short* erow = embbf + (long)n * 256;
    v8f acc;
#pragma unroll
    for (int i = 0; i < 8; ++i) acc[i] = 0.0f;
#pragma unroll
    for (int ks = 0; ks < 8; ++ks) {
      const v16bf bfrag = frag_from(erow + ks * 32, kbase);
      acc = __builtin_amdgcn_wmma_f32_16x16x32_bf16(
          false, afrag[ks], false, bfrag, (short)0, acc, false, false);
    }
    const float en = enorm[n];
#pragma unroll
    for (int i = 0; i < 8; ++i) {
      const float s = acc[i] - en;
      if (s > bestv[i] || (s == bestv[i] && n < besti[i])) { bestv[i] = s; besti[i] = n; }
    }
  }

#pragma unroll
  for (int i = 0; i < 8; ++i) {
    float bv = bestv[i]; int bi = besti[i];
#pragma unroll
    for (int off = 1; off < 16; off <<= 1) {
      const float ov = __shfl_xor(bv, off, 32);
      const int   oi = __shfl_xor(bi, off, 32);
      if (ov > bv || (ov == bv && oi < bi)) { bv = ov; bi = oi; }
    }
    if (nl == 0) {
      const int m = ((lane & 16) ? 8 : 0) + i;
      idx[p0 + m] = bi;
      atomicAdd(&hist[bi], 1);
    }
  }
}

// ---------------------------------------------------------------------------
// z_q gather (forward value of straight-through estimator == z_q).
// ---------------------------------------------------------------------------
__global__ __launch_bounds__(256) void zq_gather_kernel(
    const int* __restrict__ idx, const float* __restrict__ emb,
    float* __restrict__ zq_out, unsigned short* __restrict__ decin, long total)
{
  const long gid = (long)blockIdx.x * blockDim.x + threadIdx.x;
  if (gid >= total) return;
  const int p = (int)(gid >> 8);
  const int d = (int)(gid & 255);
  const int k = idx[p];
  const float v = emb[(long)k * 256 + d];
  zq_out[gid] = v;
  decin[gid]  = f2bfbits(v);
}

__global__ __launch_bounds__(512) void perplexity_kernel(
    const int* __restrict__ hist, float* __restrict__ out, float invN)
{
  __shared__ float sm[512];
  const int k = threadIdx.x;
  const float p = (float)hist[k] * invN;
  sm[k] = p * __logf(p + 1e-10f);
  __syncthreads();
  for (int s = 256; s > 0; s >>= 1) {
    if (k < s) sm[k] += sm[k + s];
    __syncthreads();
  }
  if (k == 0) *out = __expf(-sm[0]);
}

// ---------------------------------------------------------------------------
extern "C" void kernel_launch(void* const* d_in, const int* in_sizes, int n_in,
                              void* d_out, int out_size, void* d_ws, size_t ws_size,
                              hipStream_t stream)
{
  const float* x       = (const float*)d_in[0];
  const float* w_enc1  = (const float*)d_in[1];
  const float* b_enc1  = (const float*)d_in[2];
  const float* w_enc2  = (const float*)d_in[3];
  const float* b_enc2  = (const float*)d_in[4];
  const float* w_prevq = (const float*)d_in[5];
  const float* b_prevq = (const float*)d_in[6];
  const float* emb     = (const float*)d_in[7];
  const float* w_post  = (const float*)d_in[8];
  const float* b_post  = (const float*)d_in[9];
  const float* w_dec1  = (const float*)d_in[10];
  const float* b_dec1  = (const float*)d_in[11];
  const float* w_dec2  = (const float*)d_in[12];
  const float* b_dec2  = (const float*)d_in[13];

  // Output tuple: (x_recon, z_e, z_q, perplexity) flat-concatenated.
  float* out  = (float*)d_out;
  float* xrec = out;                       // 16*3*256*256   = 3,145,728
  float* ze   = out + 3145728L;            // 16*64*64*256   = 16,777,216
  float* zq   = ze + 16777216L;            // 16*64*64*256   = 16,777,216
  float* perp = zq + 16777216L;            // 1

  // Workspace carve-up.
  char* ws = (char*)d_ws;
  size_t off = 0;
  auto take = [&](size_t bytes) -> char* {
    char* p = ws + off;
    off = (off + bytes + 255) & ~(size_t)255;
    return p;
  };
  unsigned short* h1    = (unsigned short*)take(2UL * 16 * 128 * 128 * 64);   // enc1 out
  unsigned short* h2    = (unsigned short*)take(2UL * 16 * 64 * 64 * 128);    // enc2 out
  unsigned short* zbf   = (unsigned short*)take(2UL * 16 * 64 * 64 * 256);    // z (bf16)
  unsigned short* decin = (unsigned short*)take(2UL * 16 * 64 * 64 * 256);    // z_q (bf16)
  unsigned short* postb = (unsigned short*)take(2UL * 16 * 128 * 128 * 128);  // post out
  unsigned short* d1b   = (unsigned short*)take(2UL * 16 * 256 * 256 * 64);   // dec1 out
  unsigned short* embbf = (unsigned short*)take(2UL * 512 * 256);
  unsigned short* wbe2  = (unsigned short*)take(2UL * 128 * 1024);            // enc2 packed
  unsigned short* wbpq  = (unsigned short*)take(2UL * 256 * 1152);            // prevq packed
  unsigned short* wbpo  = (unsigned short*)take(2UL * 128 * 4096);            // post packed
  unsigned short* wbd1  = (unsigned short*)take(2UL * 64 * 2048);             // dec1 packed
  unsigned short* wbd2  = (unsigned short*)take(2UL * 16 * 576);              // dec2 packed (padded N)
  float* enorm = (float*)take(4UL * 512);
  int*   idx   = (int*)take(4UL * 65536);
  int*   hist  = (int*)take(4UL * 512);
  (void)in_sizes; (void)n_in; (void)out_size; (void)ws_size; (void)off;

  auto blocks_for = [](long waves) { return (unsigned)((waves + 7) / 8); };  // 8 waves/block
  auto blocks_elems = [](long n) { return (unsigned)((n + 255) / 256); };

  // --- weight prepack (tiny) ---
  pack_w_kernel<64, 128, 4, 4, false><<<blocks_elems(128L * 1024), 256, 0, stream>>>(w_enc2, wbe2);
  pack_w_kernel<128, 256, 3, 3, false><<<blocks_elems(256L * 1152), 256, 0, stream>>>(w_prevq, wbpq);
  pack_w_kernel<256, 128, 4, 4, true><<<blocks_elems(128L * 4096), 256, 0, stream>>>(w_post, wbpo);
  pack_w_kernel<128, 64, 4, 4, true><<<blocks_elems(64L * 2048), 256, 0, stream>>>(w_dec1, wbd1);
  pack_w_kernel<64, 3, 3, 3, true><<<blocks_elems(16L * 576), 256, 0, stream>>>(w_dec2, wbd2);

  // --- enc1: conv 3->64, k4 s2 p1, 256->128, ReLU (f32 NCHW in, generic path) ---
  conv_wmma_gen_kernel<16, 3, 64, 4, 4, 2, 1, 256, 256, 128, 128, 4, true>
      <<<blocks_for(16384L), 256, 0, stream>>>(x, w_enc1, b_enc1, h1);

  // Exact grids for staged kernels: (MGROUPS/8) * NGRP blocks.
  // --- enc2: conv 64->128, k4 s2 p1, 128->64, ReLU; MGROUPS=2048, NGRP=2 ---
  conv_wmma_fast_kernel<16, 64, 128, 4, 4, 2, 1, false, 128, 128, 64, 64, 2, 4, true, 0>
      <<<(2048 / 8) * 2, 256, 0, stream>>>(h1, wbe2, b_enc2, h2, (float*)nullptr);

  // --- prevq: conv 128->256, k3 s1 p1, 64->64; z_e f32 NHWC + z bf16; MGROUPS=2048, NGRP=4 ---
  conv_wmma_fast_kernel<16, 128, 256, 3, 3, 1, 1, false, 64, 64, 64, 64, 2, 4, false, 1>
      <<<(2048 / 8) * 4, 256, 0, stream>>>(h2, wbpq, b_prevq, zbf, ze);

  // --- VQ ---
  prep_emb_kernel<<<2, 256, 0, stream>>>(emb, embbf, enorm, hist);
  vq_argmin_kernel<<<blocks_for(4096L), 256, 0, stream>>>(zbf, embbf, enorm, idx, hist, 65536);
  zq_gather_kernel<<<65536, 256, 0, stream>>>(idx, emb, zq, decin, 16777216L);
  perplexity_kernel<<<1, 512, 0, stream>>>(hist, perp, 1.0f / 65536.0f);

  // --- post: convT 256->128, k4 s2 p1, 64->128; MGROUPS=8192, NGRP=2 ---
  conv_wmma_fast_kernel<16, 256, 128, 4, 4, 2, 1, true, 64, 64, 128, 128, 2, 4, false, 0>
      <<<(8192 / 8) * 2, 256, 0, stream>>>(decin, wbpo, b_post, postb, (float*)nullptr);

  // --- dec1: convT 128->64, k4 s2 p1, 128->256, ReLU; MGROUPS=32768, NGRP=1 ---
  conv_wmma_fast_kernel<16, 128, 64, 4, 4, 2, 1, true, 128, 128, 256, 256, 2, 4, true, 0>
      <<<(32768 / 8) * 1, 256, 0, stream>>>(postb, wbd1, b_dec1, d1b, (float*)nullptr);

  // --- dec2: convT 64->3, k3 s1 p1, 256->256; x_recon f32 NCHW; MGROUPS=32768, NGRP=1 ---
  conv_wmma_fast_kernel<16, 64, 3, 3, 3, 1, 1, true, 256, 256, 256, 256, 2, 1, false, 2>
      <<<(32768 / 8) * 1, 256, 0, stream>>>(d1b, wbd2, b_dec2, (unsigned short*)nullptr, xrec);
}
